// DeformHead_9921374453886
// MI455X (gfx1250) — matrix operand order repaired and checked by reading, not compiled, not run on hardware
//
#include <hip/hip_runtime.h>

typedef __attribute__((ext_vector_type(16))) __bf16 v16bf;
typedef __attribute__((ext_vector_type(8)))  float  v8f;
typedef __attribute__((ext_vector_type(4)))  unsigned int u32x4;
typedef __attribute__((ext_vector_type(4)))  int          i32x4;
typedef __attribute__((ext_vector_type(8)))  int          i32x8;

#define N_TOT   16384      // B*H*W = 4*64*64
#define HW      4096
#define BM      64
#define BN      128
#define BK      32
#define LDA     40         // padded LDS stride (bf16) for A tile: 16 dw data + 4 dw pad
#define LDB     34         // padded LDS stride (bf16) for B tile [128 x 32] (n-major)
#define F_PRE_RELU  1
#define F_POST_RELU 2
#define BN_INV_SQRT 0.9999950000374997f   // 1/sqrt(1 + 1e-5)

#if __has_builtin(__builtin_amdgcn_tensor_load_to_lds)
#define HAVE_TDM 1
#endif

union Frag { unsigned int u[8]; v16bf v; };

__device__ __forceinline__ void wait_tensorcnt0() {
#if __has_builtin(__builtin_amdgcn_s_wait_tensorcnt)
  __builtin_amdgcn_s_wait_tensorcnt(0);
#else
  asm volatile("s_wait_tensorcnt 0x0" ::: "memory");
#endif
}

// ---------------- TDM: DMA one [rem_m x 32] bf16 tile (row stride K) into padded LDS ----------
// pad_interval=3 (16 DWORDs) + pad_amount=3 (4 DWORDs) => LDS row stride 20 dw = 40 bf16 = LDA.
// OOB rows/cols (vs rem_m/rem_k) read as zero per TDM spec -> no tail guards needed.
__device__ __forceinline__ void tdm_load_A(const __bf16* tile, unsigned ldsOff,
                                           int rem_k, int rem_m, int K) {
#ifdef HAVE_TDM
  const unsigned long long ga = (unsigned long long)(size_t)tile;
  u32x4 g0;
  g0[0] = 1u;                                              // count=1 (valid D#)
  g0[1] = ldsOff;                                          // lds_addr (bytes)
  g0[2] = (unsigned)(ga & 0xFFFFFFFFu);                    // global_addr[31:0]
  g0[3] = (unsigned)((ga >> 32) & 0x1FFFFFFu) | (2u << 30);// global_addr[56:32] | type=2
  i32x8 g1;
  g1[0] = (1 << 16) | (1 << 20) | (3 << 22) | (3 << 25);   // data_size=2B, pad_en, intvl=16dw, amt=4dw
  g1[1] = (int)(((unsigned)rem_k & 0xFFFFu) << 16);        // tensor_dim0[15:0]
  g1[2] = (int)(((unsigned)rem_k >> 16) |
                (((unsigned)rem_m & 0xFFFFu) << 16));      // tensor_dim0[31:16] | tensor_dim1[15:0]
  g1[3] = (int)((((unsigned)rem_m >> 16) & 0xFFFFu) |
                (32u << 16));                              // tensor_dim1[31:16] | tile_dim0=32
  g1[4] = BM;                                              // tile_dim1 = 64 rows (tile_dim2=0)
  g1[5] = K;                                               // tensor_dim0_stride[31:0]
  g1[6] = 0;
  g1[7] = 0;
  const i32x4 z4 = {0, 0, 0, 0};
#if defined(__clang_major__) && (__clang_major__ >= 23)
  const i32x8 z8 = {0, 0, 0, 0, 0, 0, 0, 0};
  __builtin_amdgcn_tensor_load_to_lds(g0, g1, z4, z4, z8, 0);
#else
  __builtin_amdgcn_tensor_load_to_lds(g0, g1, z4, z4, 0);
#endif
#else
  (void)tile; (void)ldsOff; (void)rem_k; (void)rem_m; (void)K;
#endif
}

// ---------------- fallback cooperative A-tile loader ----------------
__device__ __forceinline__ void load_A_tile(const __bf16* __restrict__ A, __bf16* ldsA,
                                            int t, int mBase, int k0, int M, int K) {
  const int row = t >> 2;
  const int kc  = (t & 3) << 3;
  const int gr  = mBase + row;
  uint4 d; d.x = d.y = d.z = d.w = 0u;
  if (gr < M) {
    d = *(const uint4*)(A + (size_t)gr * K + k0 + kc);
    __builtin_prefetch(A + (size_t)gr * K + k0 + kc + BK, 0, 0);
  }
  unsigned int* dst = (unsigned int*)(ldsA + row * LDA + kc);
  dst[0] = d.x; dst[1] = d.y; dst[2] = d.z; dst[3] = d.w;
}

// ---------------- WMMA core: load fragments from LDS in contiguous k order (k = 16*h + s), 4x mma ----------------
__device__ __forceinline__ void frags_mma(const __bf16* ldsA, const __bf16* ldsB,
                                          int lane, int wm, int wn, v8f acc[2][2]) {
  const int l16 = lane & 15;
  const int hh  = lane >> 4;
  Frag a0, a1, b0, b1;
#pragma unroll
  for (int j = 0; j < 8; ++j) {
    const int kbA = (j << 1) + (hh << 4);
    a0.u[j] = *(const unsigned int*)(ldsA + (wm * 32 +      l16) * LDA + kbA);
    a1.u[j] = *(const unsigned int*)(ldsA + (wm * 32 + 16 + l16) * LDA + kbA);
    const int kbB = (j << 1) + (hh << 4);
    b0.u[j] = *(const unsigned int*)(ldsB + (wn * 32 +      l16) * LDB + kbB);
    b1.u[j] = *(const unsigned int*)(ldsB + (wn * 32 + 16 + l16) * LDB + kbB);
  }
  acc[0][0] = __builtin_amdgcn_wmma_f32_16x16x32_bf16(false, a0.v, false, b0.v, (short)0, acc[0][0], false, false);
  acc[0][1] = __builtin_amdgcn_wmma_f32_16x16x32_bf16(false, a0.v, false, b1.v, (short)0, acc[0][1], false, false);
  acc[1][0] = __builtin_amdgcn_wmma_f32_16x16x32_bf16(false, a1.v, false, b0.v, (short)0, acc[1][0], false, false);
  acc[1][1] = __builtin_amdgcn_wmma_f32_16x16x32_bf16(false, a1.v, false, b1.v, (short)0, acc[1][1], false, false);
}

__device__ __forceinline__ void epilogue(v8f acc[2][2], int mBase, int nBase, int lane,
                                         int wm, int wn,
                                         const float* bias, const float* g, const float* bb,
                                         const __bf16* res, __bf16* obf, float* of,
                                         int M, int flags) {
  const int l16 = lane & 15, hh = lane >> 4;
#pragma unroll
  for (int ai = 0; ai < 2; ++ai)
#pragma unroll
    for (int bi = 0; bi < 2; ++bi) {
      const int col = nBase + wn * 32 + bi * 16 + l16;
#pragma unroll
      for (int r = 0; r < 8; ++r) {
        const int row = mBase + wm * 32 + ai * 16 + r + 8 * hh;   // C/D layout
        if (row >= M) continue;
        float v = acc[ai][bi][r];
        if (bias) v += bias[row];
        if (flags & F_PRE_RELU) v = fmaxf(v, 0.0f);
        if (g)    v = v * (g[row] * BN_INV_SQRT) + bb[row];
        if (res)  v += (float)res[(size_t)row * N_TOT + col];
        if (flags & F_POST_RELU) v = fmaxf(v, 0.0f);
        if (obf) obf[(size_t)row * N_TOT + col] = (__bf16)v;
        if (of) {
          const int b = col >> 12, hw = col & (HW - 1);
          of[((size_t)b * M + row) * HW + hw] = v;   // NCHW fp32
        }
      }
    }
}

// ---------------- generic conv-as-GEMM (ks = 1 or 3, implicit im2col) ----------------
__global__ __launch_bounds__(256) void conv_gemm_wmma(
    const __bf16* __restrict__ A, const __bf16* __restrict__ Bact,
    const float* __restrict__ bias, const float* __restrict__ g,
    const float* __restrict__ bb, const __bf16* __restrict__ res,
    __bf16* __restrict__ obf, float* __restrict__ of,
    int M, int K, int ks, int flags) {
  __shared__ __bf16 ldsA[BM * LDA];
  __shared__ __bf16 ldsB[BN * LDB];
  const int t = threadIdx.x;
  const int nBase = blockIdx.x * BN;
  const int mBase = blockIdx.y * BM;
  const int lane = t & 31, wave = t >> 5;
  const int wm = wave >> 2, wn = wave & 3;
#ifdef HAVE_TDM
  const unsigned ldsA_off = (unsigned)(unsigned long long)(size_t)&ldsA[0];
#endif
  v8f acc[2][2] = {};
  for (int k0 = 0; k0 < K; k0 += BK) {
#ifdef HAVE_TDM
    if (wave == 0) {
      tdm_load_A(A + (size_t)mBase * K + k0, ldsA_off, K - k0, M - mBase, K);
      wait_tensorcnt0();
    }
#else
    load_A_tile(A, ldsA, t, mBase, k0, M, K);
#endif
    {
      const int krow = t >> 3;            // 0..31
      const int nc0  = (t & 7) << 4;      // 16 cols per thread
      const int k = k0 + krow;
      int c, dh, dw;
      if (ks == 1) { c = k; dh = 0; dw = 0; }
      else { c = k / 9; const int r = k - c * 9; dh = r / 3 - 1; dw = r - (r / 3) * 3 - 1; }
      const __bf16* src = Bact + (size_t)c * N_TOT;
      const int doff = dh * 64 + dw;
#pragma unroll
      for (int i = 0; i < 16; ++i) {
        const int n = nBase + nc0 + i;
        const int h = (n >> 6) & 63, w = n & 63;
        __bf16 v = (__bf16)0.0f;
        if ((unsigned)(h + dh) < 64u && (unsigned)(w + dw) < 64u) v = src[n + doff];
        ldsB[(nc0 + i) * LDB + krow] = v;
      }
    }
    __syncthreads();
    frags_mma(ldsA, ldsB, lane, wm, wn, acc);
    __syncthreads();
  }
  epilogue(acc, mBase, nBase, lane, wm, wn, bias, g, bb, res, obf, of, M, flags);
}

// ---------------- deformable GEMM: B tile gathered via precomputed bilinear meta ----------------
__global__ __launch_bounds__(256) void deform_gemm_wmma(
    const __bf16* __restrict__ A, const __bf16* __restrict__ Hbf,
    const int4* __restrict__ midx, const float4* __restrict__ mwt,
    const float* __restrict__ g, const float* __restrict__ bb,
    float* __restrict__ of, int M, int K, int flags) {
  __shared__ __bf16 ldsA[BM * LDA];
  __shared__ __bf16 ldsB[BN * LDB];
  const int t = threadIdx.x;
  const int nBase = blockIdx.x * BN;
  const int mBase = blockIdx.y * BM;
  const int lane = t & 31, wave = t >> 5;
  const int wm = wave >> 2, wn = wave & 3;
#ifdef HAVE_TDM
  const unsigned ldsA_off = (unsigned)(unsigned long long)(size_t)&ldsA[0];
#endif
  v8f acc[2][2] = {};
  for (int k0 = 0; k0 < K; k0 += BK) {
#ifdef HAVE_TDM
    if (wave == 0) {
      tdm_load_A(A + (size_t)mBase * K + k0, ldsA_off, K - k0, M - mBase, K);
      wait_tensorcnt0();
    }
#else
    load_A_tile(A, ldsA, t, mBase, k0, M, K);
#endif
    {
      const int krow = t >> 3;
      const int nc0  = (t & 7) << 4;
      const int k = k0 + krow;
      const int c = k / 9;
      const int j = k - c * 9;
      const __bf16* src = Hbf + (size_t)c * N_TOT;
#pragma unroll
      for (int i = 0; i < 16; ++i) {
        const int n = nBase + nc0 + i;
        const int4   I = midx[(size_t)n * 9 + j];
        const float4 W = mwt [(size_t)n * 9 + j];
        const float v = W.x * (float)src[I.x] + W.y * (float)src[I.y] +
                        W.z * (float)src[I.z] + W.w * (float)src[I.w];
        ldsB[(nc0 + i) * LDB + krow] = (__bf16)v;
      }
    }
    __syncthreads();
    frags_mma(ldsA, ldsB, lane, wm, wn, acc);
    __syncthreads();
  }
  epilogue(acc, mBase, nBase, lane, wm, wn, nullptr, g, bb, nullptr, nullptr, of, M, flags);
}

// ---------------- bilinear sampling metadata (ref deform_conv semantics, pad=1, ks=3) ----------------
__global__ __launch_bounds__(256) void build_meta(const float* __restrict__ off,
                                                  int4* __restrict__ midx,
                                                  float4* __restrict__ mwt) {
  const int id = blockIdx.x * blockDim.x + threadIdx.x;
  if (id >= N_TOT * 9) return;
  const int n = id / 9, j = id - n * 9;
  const int b = n >> 12, hw = n & (HW - 1), h = hw >> 6, w = hw & 63;
  const float ox = off[((size_t)b * 18 + j) * HW + hw];
  const float oy = off[((size_t)b * 18 + j + 9) * HW + hw];
  const float px = ox + (float)(j / 3 - 1) + (float)(h + 1);
  const float py = oy + (float)(j % 3 - 1) + (float)(w + 1);
  const float fx = floorf(px), fy = floorf(py);
  const float LIM = 65.0f;                     // Hp-1 = Wp-1 = 65
  const float qxl = fminf(fmaxf(fx, 0.f), LIM);
  const float qyl = fminf(fmaxf(fy, 0.f), LIM);
  const float qxr = fminf(fmaxf(fx + 1.f, 0.f), LIM);
  const float qyr = fminf(fmaxf(fy + 1.f, 0.f), LIM);
  const bool mx = (px < 1.f) || (px > 64.f);
  const bool my = (py < 1.f) || (py > 64.f);
  const float cpx = fminf(fmaxf(mx ? fx : px, 0.f), LIM);
  const float cpy = fminf(fmaxf(my ? fy : py, 0.f), LIM);
  const float glt = (1.f + (qxl - cpx)) * (1.f + (qyl - cpy));
  const float grb = (1.f - (qxr - cpx)) * (1.f - (qyr - cpy));
  const float glb = (1.f + (qxl - cpx)) * (1.f - (qyr - cpy));
  const float grt = (1.f - (qxr - cpx)) * (1.f + (qyl - cpy));
  int4 I; float4 W;
  auto mk = [&](float qx, float qy, float gv, int& oi, float& ow) {
    const int ix = (int)qx - 1, iy = (int)qy - 1;   // padded -> unpadded
    if ((unsigned)ix < 64u && (unsigned)iy < 64u) { oi = (b << 12) + (ix << 6) + iy; ow = gv; }
    else { oi = 0; ow = 0.f; }
  };
  mk(qxl, qyl, glt, I.x, W.x);
  mk(qxr, qyr, grb, I.y, W.y);
  mk(qxl, qyr, glb, I.z, W.z);
  mk(qxr, qyl, grt, I.w, W.w);
  midx[id] = I; mwt[id] = W;
}

// ---------------- fp32 -> bf16 converters ----------------
__global__ __launch_bounds__(256) void cvt_flat(const float* __restrict__ in,
                                                __bf16* __restrict__ out, int n) {
  const int i = blockIdx.x * blockDim.x + threadIdx.x;
  if (i < n) out[i] = (__bf16)in[i];
}
// NCHW fp32 -> [C][B*H*W] bf16
__global__ __launch_bounds__(256) void cvt_nchw_cn(const float* __restrict__ in,
                                                   __bf16* __restrict__ out, int C) {
  const int id = blockIdx.x * blockDim.x + threadIdx.x;
  if (id >= C * N_TOT) return;
  const int c = id >> 14, n = id & (N_TOT - 1);
  const int b = n >> 12, hw = n & (HW - 1);
  out[id] = (__bf16)in[((size_t)b * C + c) * HW + hw];
}

extern "C" void kernel_launch(void* const* d_in, const int* in_sizes, int n_in,
                              void* d_out, int out_size, void* d_ws, size_t ws_size,
                              hipStream_t stream) {
  (void)in_sizes; (void)n_in; (void)out_size; (void)ws_size;
  auto F = [&](int i) { return (const float*)d_in[i]; };

  // ---- carve workspace ----
  char* p = (char*)d_ws;
  auto carve = [&](size_t bytes) { void* r = (void*)p; p += (bytes + 255) & ~(size_t)255; return r; };
  __bf16* xbf  = (__bf16*)carve((size_t)512 * N_TOT * 2);
  __bf16* bufT = (__bf16*)carve((size_t)256 * N_TOT * 2);
  __bf16* bufU = (__bf16*)carve((size_t)256 * N_TOT * 2);
  __bf16* bufR1= (__bf16*)carve((size_t)256 * N_TOT * 2);
  __bf16* bufR2= (__bf16*)carve((size_t)256 * N_TOT * 2);
  __bf16* bufH = (__bf16*)carve((size_t)256 * N_TOT * 2);
  float*  offs = (float*)carve((size_t)18 * N_TOT * 4);
  int4*   midx = (int4*)carve((size_t)N_TOT * 9 * 16);
  float4* mwt  = (float4*)carve((size_t)N_TOT * 9 * 16);

  // ---- weight tensors: input index + element count (insertion order of setup_inputs) ----
  struct WSpec { int idx; int elems; };
  const WSpec wl[17] = {
      {1, 256 * 512},
      {3, 65536}, {7, 589824}, {11, 65536},
      {15, 65536}, {19, 589824}, {23, 65536},
      {27, 65536},
      {31, 65536},
      {33, 65536}, {37, 589824}, {41, 65536},
      {45, 65536}, {49, 589824}, {53, 65536},
      {57, 18 * 256},
      {59, 589824}
  };
  __bf16* Wbf[17];
  for (int i = 0; i < 17; ++i) {
    Wbf[i] = (__bf16*)carve((size_t)wl[i].elems * 2);
    cvt_flat<<<(wl[i].elems + 255) / 256, 256, 0, stream>>>(F(wl[i].idx), Wbf[i], wl[i].elems);
  }
  cvt_nchw_cn<<<(512 * N_TOT + 255) / 256, 256, 0, stream>>>(F(0), xbf, 512);

  auto run = [&](const __bf16* A, const __bf16* B, const float* bias, const float* g,
                 const float* bb, const __bf16* res, __bf16* obf, float* of,
                 int M, int K, int ks, int flags) {
    dim3 grid(N_TOT / BN, (M + BM - 1) / BM);
    conv_gemm_wmma<<<grid, 256, 0, stream>>>(A, B, bias, g, bb, res, obf, of, M, K, ks, flags);
  };
  auto resblock = [&](const __bf16* in, __bf16* out, int pb,
                      const __bf16* W1, const __bf16* W2, const __bf16* W3) {
    run(W1, in,    F(pb + 1), F(pb + 2),  F(pb + 3),  nullptr, bufR1, nullptr, 256, 256, 1, F_POST_RELU);
    run(W2, bufR1, F(pb + 5), F(pb + 6),  F(pb + 7),  nullptr, bufR2, nullptr, 256, 2304, 3, F_POST_RELU);
    run(W3, bufR2, F(pb + 9), F(pb + 10), F(pb + 11), in,      out,   nullptr, 256, 256, 1, F_POST_RELU);
  };

  // ---- trunk ----
  run(Wbf[0], xbf, F(2), nullptr, nullptr, nullptr, bufT, nullptr, 256, 512, 1, 0);
  resblock(bufT, bufU, 3,  Wbf[1], Wbf[2], Wbf[3]);
  resblock(bufU, bufT, 15, Wbf[4], Wbf[5], Wbf[6]);
  run(Wbf[7], bufT, F(28), F(29), F(30), nullptr, bufH, nullptr, 256, 256, 1, F_PRE_RELU);

  // ---- offset branch ----
  run(Wbf[8], bufH, F(32), nullptr, nullptr, nullptr, bufT, nullptr, 256, 256, 1, 0);
  resblock(bufT, bufU, 33, Wbf[9],  Wbf[10], Wbf[11]);
  resblock(bufU, bufT, 45, Wbf[12], Wbf[13], Wbf[14]);
  run(Wbf[15], bufT, F(58), nullptr, nullptr, nullptr, nullptr, offs, 18, 256, 1, 0);

  // ---- deformable sampling metadata + final GEMM (relu then deform_bn) ----
  build_meta<<<(N_TOT * 9 + 255) / 256, 256, 0, stream>>>(offs, midx, mwt);
  dim3 dgrid(N_TOT / BN, 256 / BM);
  deform_gemm_wmma<<<dgrid, 256, 0, stream>>>(Wbf[16], bufH, midx, mwt,
                                              F(60), F(61), (float*)d_out,
                                              256, 2304, F_PRE_RELU);
}
